// SkipGRU_52621939311120
// MI455X (gfx1250) — matrix operand order repaired and checked by reading, not compiled, not run on hardware
//
#include <hip/hip_runtime.h>

// ---------------------------------------------------------------------------
// SkipGRU on MI455X (gfx1250, wave32, WMMA)
// B=64, T=4096, F=U=128, P=16  ->  1024 independent chains of length 256.
// One persistent workgroup per batch element: its 16 chains are one WMMA
// M-tile whose per-step inputs/outputs are contiguous 16x128 blocks.
// bf16 WMMA (v_wmma_f32_16x16x32_bf16) with f32 accumulation; both weight
// matrices live in LDS as pre-packed B-fragments (192 KB of the 320 KB WGP
// LDS); recurrent-weight fragments additionally cached in VGPRs.
// ---------------------------------------------------------------------------

typedef __bf16 bf16_t;
typedef __attribute__((ext_vector_type(16))) __bf16 v16bf;
typedef __attribute__((ext_vector_type(8)))  float  v8f;

#define Bb   64
#define Tt   4096
#define Ff   128
#define Uu   128
#define Pp   16
#define Ll   256                 // T / P : sequential steps per chain
#define KS   4                   // K=128 -> 4 WMMA k-steps of 32
#define WFRAG_DW 49152           // 2 gemms * 24 tiles * 4 ks * 32 lanes * 8 dw
#define XSTR 136                 // padded bf16 row stride (272 B = 17*16 B)

__device__ __forceinline__ unsigned short f2bf(float f) {
    union { float f; unsigned u; } x; x.f = f;
    unsigned r = x.u + 0x7fffu + ((x.u >> 16) & 1u);   // round-to-nearest-even
    return (unsigned short)(r >> 16);
}
__device__ __forceinline__ unsigned pack2bf(float a, float b) {
    return (unsigned)f2bf(a) | ((unsigned)f2bf(b) << 16);
}

union ABfrag { uint4 q[2]; v16bf v; };

__global__ void __launch_bounds__(256, 1)
skipgru_persistent(const float* __restrict__ x,     // [B,T,F]
                   const float* __restrict__ wi,    // [F,3U]
                   const float* __restrict__ wr,    // [U,3U]
                   const float* __restrict__ bias,  // [2,3U]
                   float* __restrict__ out)         // [B,T,U]
{
    extern __shared__ unsigned smem[];
    unsigned* wfrag = smem;                               // 192 KB
    bf16_t*   xbf   = (bf16_t*)(smem + WFRAG_DW);         // 16 x XSTR bf16
    bf16_t*   hbf   = xbf + 16 * XSTR;                    // 16 x XSTR bf16

    const int tid  = threadIdx.x;
    const int lane = tid & 31;
    const int w    = tid >> 5;          // wave id 0..7 -> owns u cols [16w,16w+16)
    const int half = lane >> 4;
    const int n    = lane & 15;
    const int m    = lane & 15;         // A-fragment row
    const int b    = blockIdx.x;

    // ---- one-time: pack W_in / W_rec into WMMA B-fragment order (bf16) ----
    // word index widx = (((g*24 + tile)*4 + s)*32 + lane)*8 + v
    for (int i = 0; i < WFRAG_DW / 256; ++i) {
        int widx = tid + i * 256;
        int v  = widx & 7;
        int ln = (widx >> 3) & 31;
        int s  = (widx >> 8) & 3;
        int t  = (widx >> 10) % 24;
        int g  = widx / 24576;
        int hb = ln >> 4, nn = ln & 15;
        int k  = 32 * s + 16 * hb + 2 * v;          // B[k][col], B[k+1][col]
        int col = t * 16 + nn;
        const float* src = g ? wr : wi;
        wfrag[widx] = pack2bf(src[k * 384 + col], src[(k + 1) * 384 + col]);
    }
    // zero bf16 h state
    unsigned* hbf32 = (unsigned*)hbf;
    for (int i = tid; i < 16 * XSTR / 2; i += 256) hbf32[i] = 0u;
    __syncthreads();

    // ---- cache this wave's recurrent-weight fragments in VGPRs (96 regs) ----
    uint4 wrf[3][KS][2];
#pragma unroll
    for (int t3 = 0; t3 < 3; ++t3) {
        int tile = w + t3 * 8;
#pragma unroll
        for (int s = 0; s < KS; ++s) {
            const uint4* p =
                (const uint4*)&wfrag[(((24 + tile) * 4 + s) * 32 + lane) * 8];
            wrf[t3][s][0] = p[0];
            wrf[t3][s][1] = p[1];
        }
    }

    // per-column biases (broadcast over the 16 rows of each tile)
    const int u0 = w * 16 + n;
    const float bin[3] = { bias[u0], bias[128 + u0], bias[256 + u0] };
    const float brc[3] = { bias[384 + u0], bias[512 + u0], bias[640 + u0] };

    v8f hprev = {0.f, 0.f, 0.f, 0.f, 0.f, 0.f, 0.f, 0.f};
    const float* xblk = x + (size_t)b * Tt * Ff;

    for (int l = 0; l < Ll; ++l) {
        // ---- stage x_t (16 consecutive timesteps, contiguous 8 KB) -> bf16 ----
        const float* src = xblk + (size_t)l * Pp * Ff;
#pragma unroll
        for (int i = 0; i < 4; ++i) {
            int p  = tid + i * 256;          // float2 pair index, coalesced
            int mr = p >> 6;
            int k  = (p & 63) * 2;
            float2 f = *(const float2*)(src + mr * Ff + k);
            ((unsigned*)xbf)[(mr * XSTR + k) >> 1] = pack2bf(f.x, f.y);
        }
        if (l + 1 < Ll)                       // gfx1250 global_prefetch next block
            __builtin_prefetch(src + Pp * Ff + tid * 8, 0, 1);
        __syncthreads();                      // xbf + hbf writes visible

        // ---- dual GEMM: acc = bias; acc += A * B over 4 k-steps ----
        v8f accX[3], accH[3];
#pragma unroll
        for (int t3 = 0; t3 < 3; ++t3)
#pragma unroll
            for (int i = 0; i < 8; ++i) { accX[t3][i] = bin[t3]; accH[t3][i] = brc[t3]; }

#pragma unroll
        for (int s = 0; s < KS; ++s) {
            ABfrag ax, ah;
            const bf16_t* pax = xbf + m * XSTR + 32 * s + 16 * half;
            const bf16_t* pah = hbf + m * XSTR + 32 * s + 16 * half;
            ax.q[0] = *(const uint4*)(pax);      ax.q[1] = *(const uint4*)(pax + 8);
            ah.q[0] = *(const uint4*)(pah);      ah.q[1] = *(const uint4*)(pah + 8);
#pragma unroll
            for (int t3 = 0; t3 < 3; ++t3) {
                int tile = w + t3 * 8;
                ABfrag bw;
                const uint4* p =
                    (const uint4*)&wfrag[((tile * 4 + s) * 32 + lane) * 8];
                bw.q[0] = p[0]; bw.q[1] = p[1];
                accX[t3] = __builtin_amdgcn_wmma_f32_16x16x32_bf16(
                    false, ax.v, false, bw.v, (short)0, accX[t3], false, false);
                ABfrag br; br.q[0] = wrf[t3][s][0]; br.q[1] = wrf[t3][s][1];
                accH[t3] = __builtin_amdgcn_wmma_f32_16x16x32_bf16(
                    false, ah.v, false, br.v, (short)0, accH[t3], false, false);
            }
        }
        __syncthreads();                      // all LDS reads done before h update

        // ---- gates + state update (D-fragment positions identical each step) ----
        const size_t orow = ((size_t)b * Tt + (size_t)l * Pp) * Uu;
#pragma unroll
        for (int i = 0; i < 8; ++i) {
            int mr  = i + 8 * half;           // D layout: M = vgpr + 8*(lane/16)
            float z = 1.f / (1.f + __expf(-(accX[0][i] + accH[0][i])));
            float r = 1.f / (1.f + __expf(-(accX[1][i] + accH[1][i])));
            float hh = accX[2][i] + r * accH[2][i];
            hh = hh > 0.f ? hh : 0.f;         // activation='relu'
            float hn = z * hprev[i] + (1.f - z) * hh;
            hprev[i] = hn;
            out[orow + (size_t)mr * Uu + u0] = hn;
            ((unsigned short*)hbf)[mr * XSTR + u0] = f2bf(hn);
        }
        // next top-of-loop __syncthreads() fences hbf stores vs. next GEMM reads
    }
}

extern "C" void kernel_launch(void* const* d_in, const int* in_sizes, int n_in,
                              void* d_out, int out_size, void* d_ws, size_t ws_size,
                              hipStream_t stream) {
    (void)in_sizes; (void)n_in; (void)d_ws; (void)ws_size; (void)out_size;
    const float* x    = (const float*)d_in[0];
    const float* wi   = (const float*)d_in[1];
    const float* wr   = (const float*)d_in[2];
    const float* bias = (const float*)d_in[3];
    float* out = (float*)d_out;

    const size_t shmem = (size_t)WFRAG_DW * sizeof(unsigned)   // 192 KB weights
                       + 2u * 16u * XSTR * sizeof(bf16_t);     // x + h staging
    skipgru_persistent<<<dim3(Bb), dim3(256), shmem, stream>>>(x, wi, wr, bias, out);
}
